// CrossChannelAttention_6365141533154
// MI455X (gfx1250) — matrix operand order repaired and checked by reading, not compiled, not run on hardware
//
#include <hip/hip_runtime.h>
#include <hip/hip_bf16.h>

// ---------------------------------------------------------------------------
// CrossChannelAttention for MI455X (gfx1250), wave32 + v_wmma_f32_16x16x32_bf16
// Bandwidth-bound workload: bf16 intermediates, fp32 accumulation.
// GEMM B-panels staged in LDS (320KB/WGP) to kill 8x intra-block reload;
// global_prefetch issued ahead of the attention contraction loop.
// ---------------------------------------------------------------------------

typedef __bf16 bf16_t;
typedef __attribute__((ext_vector_type(16))) __bf16 v16bf;
typedef __attribute__((ext_vector_type(8)))  __bf16 v8bf;
typedef __attribute__((ext_vector_type(4)))  __bf16 v4bf;
typedef __attribute__((ext_vector_type(8)))  float  v8f;

#define NB    8      // batch
#define CIN   256    // DIM
#define D2C   512    // DIM*2
#define HEADS 8
#define HW    4096   // 64*64

__device__ __forceinline__ bf16_t f2bf(float f) {
  unsigned u = __builtin_bit_cast(unsigned, f);
  unsigned r = u + 0x7FFFu + ((u >> 16) & 1u);   // round-to-nearest-even
  unsigned short h = (unsigned short)(r >> 16);
  return __builtin_bit_cast(bf16_t, h);
}

__device__ __forceinline__ v8f wmma_bf16(v16bf a, v16bf b, v8f c) {
  return __builtin_amdgcn_wmma_f32_16x16x32_bf16(false, a, false, b,
                                                 (short)0, c, false, false);
}

// A fragment: A row-major [*, lda]; 16x32 tile at (row0, kb).
// lane m (0..15): K 0..15; lane m+16: K 16..31 of row m.
__device__ __forceinline__ v16bf load_a_frag(const bf16_t* __restrict__ A,
                                             int lda, int row0, int kb) {
  int lane = threadIdx.x & 31;
  int m = lane & 15, hi = lane >> 4;
  const bf16_t* p = A + (size_t)(row0 + m) * lda + kb + 16 * hi;
  v8bf lo = *(const v8bf*)(p);
  v8bf hv = *(const v8bf*)(p + 8);
  return __builtin_shufflevector(lo, hv, 0,1,2,3,4,5,6,7,8,9,10,11,12,13,14,15);
}

// B fragment from transposed operand Bt row-major [N, ldb] (Bt[n][k] = B[k][n]).
// lane n (0..15): K 0..15; lane n+16: K 16..31 of column n. One 32B load.
__device__ __forceinline__ v16bf load_b_frag(const bf16_t* __restrict__ Bt,
                                             int ldb, int col0, int kb) {
  int lane = threadIdx.x & 31;
  int n = lane & 15, hi = lane >> 4;
  const bf16_t* p = Bt + (size_t)(col0 + n) * ldb + kb + 16 * hi;
  return *(const v16bf*)p;
}

// --------------------------- small converters ------------------------------

__global__ void cvt_f32_bf16_kernel(const float* __restrict__ src,
                                    bf16_t* __restrict__ dst, int n) {
  int i = blockIdx.x * blockDim.x + threadIdx.x;
  if (i < n) dst[i] = f2bf(src[i]);
}

// x: [batch][C][N] fp32 -> xt: [batch][N][C] bf16   (32x32 LDS tiles)
__global__ __launch_bounds__(256) void transpose_cvt_kernel(
    const float* __restrict__ x, bf16_t* __restrict__ xt, int C, int N) {
  __shared__ float tile[32][33];
  int b = blockIdx.z;
  int nb = blockIdx.x * 32, cb = blockIdx.y * 32;
  int t = threadIdx.x;
  int r = t >> 3, c4 = (t & 7) * 4;
  const float* xb = x + ((size_t)b * C + cb) * N + nb;
  float4 v = *(const float4*)(xb + (size_t)r * N + c4);
  tile[r][c4 + 0] = v.x; tile[r][c4 + 1] = v.y;
  tile[r][c4 + 2] = v.z; tile[r][c4 + 3] = v.w;
  __syncthreads();
  bf16_t* xtb = xt + ((size_t)b * N + nb) * C + cb;
  v4bf o = { f2bf(tile[c4 + 0][r]), f2bf(tile[c4 + 1][r]),
             f2bf(tile[c4 + 2][r]), f2bf(tile[c4 + 3][r]) };
  *(v4bf*)(xtb + (size_t)r * C + c4) = o;
}

// --------------------------- WMMA GEMM + bias ------------------------------
// out[b][m][n] (fp32) = sum_k A[m][k] * Bt[b][n][k] + bias[m]
// block = 256 thr (8 waves); block tile 128(M) x 64(N); wave tile 16(M) x 64(N)
// The 64 x K B-panel is staged once in LDS (padded rows) and shared by all
// 8 waves; A fragments stream from global (weights are L2-resident).
__global__ __launch_bounds__(256) void gemm_bias_kernel(
    const bf16_t* __restrict__ A, const bf16_t* __restrict__ Bt,
    const float* __restrict__ bias, float* __restrict__ out,
    int M, int N, int K) {
  __shared__ bf16_t Bs[64 * (512 + 8)];          // worst case K=512: 66,560 B
  int b = blockIdx.z;
  int wave = threadIdx.x >> 5, lane = threadIdx.x & 31;
  int nb = blockIdx.x * 64;
  int row0 = blockIdx.y * 128 + wave * 16;
  const bf16_t* Btb = Bt + (size_t)b * N * K;
  float* outb = out + (size_t)b * M * N;
  const int ldbs = K + 8;                        // padded LDS row stride

  // cooperative stage of B panel: 64 rows x K, 16B chunks
  int cpr = K >> 3;                              // 16B chunks per row
  int total = 64 * cpr;
  for (int c = threadIdx.x; c < total; c += 256) {
    int row = c / cpr, col = (c - row * cpr) * 8;
    *(v8bf*)&Bs[row * ldbs + col] =
        *(const v8bf*)(Btb + (size_t)(nb + row) * K + col);
  }
  __syncthreads();

  int hi = lane >> 4, n = lane & 15;
  v8f acc[4] = {};
  for (int kb = 0; kb < K; kb += 32) {
    v16bf a = load_a_frag(A, K, row0, kb);
#pragma unroll
    for (int t = 0; t < 4; ++t) {
      int base = (t * 16 + n) * ldbs + kb + 16 * hi;
      v8bf lo = *(const v8bf*)&Bs[base];
      v8bf hv = *(const v8bf*)&Bs[base + 8];
      v16bf bfr = __builtin_shufflevector(
          lo, hv, 0,1,2,3,4,5,6,7,8,9,10,11,12,13,14,15);
      acc[t] = wmma_bf16(a, bfr, acc[t]);
    }
  }
#pragma unroll
  for (int t = 0; t < 4; ++t) {
#pragma unroll
    for (int j = 0; j < 8; ++j) {
      int r = row0 + j + 8 * hi;
      outb[(size_t)r * N + nb + t * 16 + n] = acc[t][j] + bias[r];
    }
  }
}

// ------------------- depthwise 3x3 + optional L2 norm ----------------------
// one block per (b,c) 64x64 plane; mode 1: normalize -> [B][512][HW] bf16
//                                  mode 0: v path    -> Vt [B][8][HW][64] bf16
__global__ __launch_bounds__(256) void dwconv_kernel(
    const float* __restrict__ mid, const float* __restrict__ dw,
    const float* __restrict__ dwb, bf16_t* __restrict__ out, int mode) {
  int plane = blockIdx.x;            // b*512 + c
  int c = plane & 511, b = plane >> 9;
  __shared__ float sm[64 * 64];
  __shared__ float red[9];
  const float* src = mid + (size_t)plane * HW;
  int t = threadIdx.x;
#pragma unroll
  for (int i = 0; i < 16; ++i) sm[i * 256 + t] = src[i * 256 + t];
  float w9[9];
#pragma unroll
  for (int i = 0; i < 9; ++i) w9[i] = dw[c * 9 + i];
  float bias = dwb[c];
  __syncthreads();

  float r[16];
  float ssq = 0.f;
#pragma unroll
  for (int i = 0; i < 16; ++i) {
    int idx = i * 256 + t;
    int y = idx >> 6, x = idx & 63;
    float a = bias;
#pragma unroll
    for (int dy = -1; dy <= 1; ++dy) {
      int yy = y + dy;
      if (yy < 0 || yy > 63) continue;
#pragma unroll
      for (int dx = -1; dx <= 1; ++dx) {
        int xx = x + dx;
        if (xx < 0 || xx > 63) continue;
        a += w9[(dy + 1) * 3 + (dx + 1)] * sm[yy * 64 + xx];
      }
    }
    r[i] = a;
    ssq += a * a;
  }

  if (mode == 1) {                   // L2 normalize over the whole plane
    for (int off = 16; off; off >>= 1) ssq += __shfl_xor(ssq, off, 32);
    if ((t & 31) == 0) red[t >> 5] = ssq;
    __syncthreads();
    if (t == 0) {
      float s = 0.f;
      for (int i = 0; i < 8; ++i) s += red[i];
      red[8] = 1.f / fmaxf(sqrtf(s), 1e-12f);
    }
    __syncthreads();
    float scale = red[8];
    bf16_t* dst = out + (size_t)plane * HW;
#pragma unroll
    for (int i = 0; i < 16; ++i) dst[i * 256 + t] = f2bf(r[i] * scale);
  } else {                           // V: write transposed [b][h][n][d]
    int h = c >> 6, d = c & 63;
    bf16_t* dst = out + ((size_t)(b * HEADS + h) * HW) * 64 + d;
#pragma unroll
    for (int i = 0; i < 16; ++i) dst[(size_t)(i * 256 + t) * 64] = f2bf(r[i]);
  }
}

// ---------------- channel attention: S = Q K^T, softmax -> P ---------------
// one block per (b,h): 64x64 scores via 16 WMMA tiles (2 per wave), K=4096
__global__ __launch_bounds__(256) void attn_kernel(
    const bf16_t* __restrict__ Qn, const bf16_t* __restrict__ Kn,
    const float* __restrict__ temp, bf16_t* __restrict__ P) {
  int bh = blockIdx.x;               // b*8 + h
  int h = bh & 7, b = bh >> 3;
  __shared__ float S[64][66];
  const bf16_t* Qh = Qn + ((size_t)b * D2C + h * 64) * HW;
  const bf16_t* Kh = Kn + ((size_t)b * D2C + h * 64) * HW;
  int wave = threadIdx.x >> 5, lane = threadIdx.x & 31;
  int mi = wave & 3;                 // M tile
  int nj = (wave >> 2) * 2;          // two N tiles per wave (shared A frag)
  int m16 = lane & 15;

  v8f acc0 = {}, acc1 = {};
  for (int kb = 0; kb < HW; kb += 32) {
    // prefetch 16 k-steps (512 elems) ahead (global_prefetch_b8); skipped in the last 16 steps
    if (kb + 512 < HW) {
      __builtin_prefetch(Qh + (size_t)(mi * 16 + m16) * HW + kb + 512, 0, 3);
      __builtin_prefetch(Kh + (size_t)(nj * 16 + m16) * HW + kb + 512, 0, 3);
    }
    v16bf a  = load_a_frag(Qh, HW, mi * 16, kb);
    v16bf b0 = load_b_frag(Kh, HW, nj * 16, kb);
    v16bf b1 = load_b_frag(Kh, HW, (nj + 1) * 16, kb);
    acc0 = wmma_bf16(a, b0, acc0);
    acc1 = wmma_bf16(a, b1, acc1);
  }
  float tm = temp[h];
  int hi = lane >> 4, n = lane & 15;
#pragma unroll
  for (int j = 0; j < 8; ++j) {
    int r = mi * 16 + j + 8 * hi;
    S[r][nj * 16 + n]       = acc0[j] * tm;
    S[r][(nj + 1) * 16 + n] = acc1[j] * tm;
  }
  __syncthreads();
  if (threadIdx.x < 64) {            // softmax along d (row of S)
    int row = threadIdx.x;
    float mx = -1e30f;
    for (int j = 0; j < 64; ++j) mx = fmaxf(mx, S[row][j]);
    float sum = 0.f;
    for (int j = 0; j < 64; ++j) sum += __expf(S[row][j] - mx);
    float inv = 1.f / sum;
    bf16_t* dst = P + ((size_t)bh * 64 + row) * 64;
    for (int j = 0; j < 64; ++j) dst[j] = f2bf(__expf(S[row][j] - mx) * inv);
  }
}

// ---------------- out = P @ V, written transposed for final GEMM -----------
// OT[b][n][h*64 + c] = sum_d P[bh][c][d] * Vt[bh][n][d]
__global__ __launch_bounds__(256) void pv_kernel(
    const bf16_t* __restrict__ P, const bf16_t* __restrict__ Vt,
    bf16_t* __restrict__ OT) {
  int bh = blockIdx.y;
  int h = bh & 7, b = bh >> 3;
  int wave = threadIdx.x >> 5, lane = threadIdx.x & 31;
  int nb = blockIdx.x * 128 + wave * 16;    // 16 spatial cols per wave
  const bf16_t* Ph  = P  + (size_t)bh * 64 * 64;
  const bf16_t* Vth = Vt + (size_t)bh * HW * 64;

  v8f acc[4] = {};
#pragma unroll
  for (int kb = 0; kb < 64; kb += 32) {
    v16bf bfr = load_b_frag(Vth, 64, nb, kb);
#pragma unroll
    for (int mi = 0; mi < 4; ++mi) {
      v16bf a = load_a_frag(Ph, 64, mi * 16, kb);
      acc[mi] = wmma_bf16(a, bfr, acc[mi]);
    }
  }
  int hi = lane >> 4, n = lane & 15;
  bf16_t* dst = OT + ((size_t)b * HW + nb + n) * D2C + h * 64;
#pragma unroll
  for (int mi = 0; mi < 4; ++mi)
#pragma unroll
    for (int j = 0; j < 8; ++j)
      dst[mi * 16 + j + 8 * hi] = f2bf(acc[mi][j]);
}

// ---------------------------------------------------------------------------

extern "C" void kernel_launch(void* const* d_in, const int* in_sizes, int n_in,
                              void* d_out, int out_size, void* d_ws, size_t ws_size,
                              hipStream_t stream) {
  (void)in_sizes; (void)n_in; (void)out_size; (void)ws_size;
  const float* encoder = (const float*)d_in[0];
  const float* decoder = (const float*)d_in[1];
  const float* wq      = (const float*)d_in[2];
  const float* bq      = (const float*)d_in[3];
  const float* dwq     = (const float*)d_in[4];
  const float* dwq_b   = (const float*)d_in[5];
  const float* wk      = (const float*)d_in[6];
  const float* bk      = (const float*)d_in[7];
  const float* dwk     = (const float*)d_in[8];
  const float* dwk_b   = (const float*)d_in[9];
  const float* wv      = (const float*)d_in[10];
  const float* bv      = (const float*)d_in[11];
  const float* dwv     = (const float*)d_in[12];
  const float* dwv_b   = (const float*)d_in[13];
  const float* temperature = (const float*)d_in[14];
  const float* wo      = (const float*)d_in[15];
  const float* bo      = (const float*)d_in[16];
  float* out = (float*)d_out;

  char* ws = (char*)d_ws;
  size_t off = 0;
  auto take = [&](size_t bytes) {
    char* p = ws + off;
    off += (bytes + 255) & ~(size_t)255;
    return p;
  };
  bf16_t* encT = (bf16_t*)take((size_t)NB * HW * CIN * 2);   // [b][n][256]
  bf16_t* decT = (bf16_t*)take((size_t)NB * HW * CIN * 2);
  bf16_t* wqb  = (bf16_t*)take((size_t)D2C * CIN * 2);
  bf16_t* wkb  = (bf16_t*)take((size_t)D2C * CIN * 2);
  bf16_t* wvb  = (bf16_t*)take((size_t)D2C * CIN * 2);
  bf16_t* wob  = (bf16_t*)take((size_t)CIN * D2C * 2);
  float*  mid  = (float*) take((size_t)NB * D2C * HW * 4);   // reused q->k->v
  bf16_t* Qn   = (bf16_t*)take((size_t)NB * D2C * HW * 2);   // normalized
  bf16_t* Kn   = (bf16_t*)take((size_t)NB * D2C * HW * 2);
  bf16_t* Vt   = (bf16_t*)take((size_t)NB * HEADS * HW * 64 * 2);
  bf16_t* Pm   = (bf16_t*)take((size_t)NB * HEADS * 64 * 64 * 2);
  bf16_t* OT   = (bf16_t*)take((size_t)NB * HW * D2C * 2);   // [b][n][512]

  // 1) weights -> bf16
  int wn = D2C * CIN;  // 131072 (same count for wo)
  cvt_f32_bf16_kernel<<<(wn + 255) / 256, 256, 0, stream>>>(wq, wqb, wn);
  cvt_f32_bf16_kernel<<<(wn + 255) / 256, 256, 0, stream>>>(wk, wkb, wn);
  cvt_f32_bf16_kernel<<<(wn + 255) / 256, 256, 0, stream>>>(wv, wvb, wn);
  cvt_f32_bf16_kernel<<<(wn + 255) / 256, 256, 0, stream>>>(wo, wob, wn);

  // 2) activations -> bf16 transposed [b][n][c]
  dim3 tg(HW / 32, CIN / 32, NB);
  transpose_cvt_kernel<<<tg, 256, 0, stream>>>(encoder, encT, CIN, HW);
  transpose_cvt_kernel<<<tg, 256, 0, stream>>>(decoder, decT, CIN, HW);

  dim3 gg(HW / 64, D2C / 128, NB);      // conv1x1 grid (M=512)
  // 3) q path
  gemm_bias_kernel<<<gg, 256, 0, stream>>>(wqb, encT, bq, mid, D2C, HW, CIN);
  dwconv_kernel<<<NB * D2C, 256, 0, stream>>>(mid, dwq, dwq_b, Qn, 1);
  // 4) k path
  gemm_bias_kernel<<<gg, 256, 0, stream>>>(wkb, decT, bk, mid, D2C, HW, CIN);
  dwconv_kernel<<<NB * D2C, 256, 0, stream>>>(mid, dwk, dwk_b, Kn, 1);
  // 5) v path (transposed bf16 output)
  gemm_bias_kernel<<<gg, 256, 0, stream>>>(wvb, decT, bv, mid, D2C, HW, CIN);
  dwconv_kernel<<<NB * D2C, 256, 0, stream>>>(mid, dwv, dwv_b, Vt, 0);

  // 6) channel attention scores + softmax
  attn_kernel<<<NB * HEADS, 256, 0, stream>>>(Qn, Kn, temperature, Pm);

  // 7) P @ V -> OT [b][n][512]
  dim3 pg(HW / 128, NB * HEADS);
  pv_kernel<<<pg, 256, 0, stream>>>(Pm, Vt, OT);

  // 8) final conv1x1: wo[256,512] @ OT -> out fp32 [b][256][hw]
  dim3 fg(HW / 64, CIN / 128, NB);
  gemm_bias_kernel<<<fg, 256, 0, stream>>>(wob, OT, bo, out, CIN, HW, D2C);
}
